// TransformerLayer_37692632990169
// MI455X (gfx1250) — matrix operand order repaired and checked by reading, not compiled, not run on hardware
//
#include <hip/hip_runtime.h>
#include <hip/hip_bf16.h>

// ---------------- CDNA5 WMMA types ----------------
typedef __bf16 bf16_t;
typedef bf16_t v16bf __attribute__((ext_vector_type(16)));
typedef bf16_t v8bf  __attribute__((ext_vector_type(8)));
typedef float  v8f   __attribute__((ext_vector_type(8)));

__device__ __forceinline__ v8f wmma_bf16(v16bf a, v16bf b, v8f c) {
  return __builtin_amdgcn_wmma_f32_16x16x32_bf16(false, a, false, b, (short)0, c, false, false);
}

// ---------------- problem constants ----------------
constexpr int E  = 1024;
constexpr int KD = 64;     // q/k head dim
constexpr int DH = 64;     // v head dim
constexpr int H  = 16;
constexpr int F  = 4096;
constexpr int Bc = 4;
constexpr int Sc = 2048;
constexpr int TOK = Bc * Sc;        // 8192
#define EPS_F 1.1920929e-07f

// ----------------------------------------------------------------------------
// Fragment helpers (wave32; contiguous k order used by this file for A and B):
// A (16x32 bf16, MxK): lane l -> row m=l&15; K run [base..base+15],
//   base = (l>=16)?16:0.
// B (32x16 bf16, KxN): lane l -> col n=l&15; K run [koff..koff+15],
//   koff = (l>=16)?16:0.
// C/D (16x16 f32): lane l -> col n=l&15; rows m(i) = ((l>=16)?8:0)+i.
// ----------------------------------------------------------------------------
__device__ __forceinline__ v16bf load_a(const bf16_t* base, int ld, int lane) {
  const bf16_t* p = base + (long)(lane & 15) * ld + ((lane >> 4) << 4);
  v8bf lo = *(const v8bf*)p;
  v8bf hi = *(const v8bf*)(p + 8);
  v16bf r;
#pragma unroll
  for (int j = 0; j < 8; ++j) { r[j] = lo[j]; r[j + 8] = hi[j]; }
  return r;
}

// B fragment from "transposed" storage: element (k,n) lives at p0[n*ld + k].
__device__ __forceinline__ v16bf load_b_trans(const bf16_t* p0, long ld, int lane) {
  return *(const v16bf*)(p0 + (long)(lane & 15) * ld + ((lane >> 4) << 4));
}

// B fragment from packed weights: 32 lanes x 16 bf16 contiguous per fragment.
__device__ __forceinline__ v16bf load_b_packed(const bf16_t* pack, long frag, int lane) {
  return *(const v16bf*)(pack + (frag * 32 + lane) * 16);
}

// ----------------------------------------------------------------------------
// f32 -> bf16 bulk convert (makes the QKV tile staging a pure byte copy).
// ----------------------------------------------------------------------------
__global__ __launch_bounds__(256) void tobf16_kernel(
    const float* __restrict__ src, bf16_t* __restrict__ dst, long n) {
  long i = ((long)blockIdx.x * 256 + threadIdx.x) * 8;
  if (i + 8 <= n) {
    float4 a = *(const float4*)(src + i);
    float4 b = *(const float4*)(src + i + 4);
    v8bf o;
    o[0] = (bf16_t)a.x; o[1] = (bf16_t)a.y; o[2] = (bf16_t)a.z; o[3] = (bf16_t)a.w;
    o[4] = (bf16_t)b.x; o[5] = (bf16_t)b.y; o[6] = (bf16_t)b.z; o[7] = (bf16_t)b.w;
    *(v8bf*)(dst + i) = o;
  }
}

// ----------------------------------------------------------------------------
// Weight repack: f32 row-major (Kdim x N) -> bf16 fragment-major.
// ----------------------------------------------------------------------------
__global__ __launch_bounds__(256) void pack_kernel(
    const float* __restrict__ src, bf16_t* __restrict__ dst,
    int Kdim, int N, long srcStride, long dstStride, float scale) {
  const int lane = threadIdx.x & 31, wave = threadIdx.x >> 5;
  const int ntiles = N >> 4;
  const long nfrags = (long)(Kdim >> 5) * ntiles;
  long frag = (long)blockIdx.x * 8 + wave;
  if (frag >= nfrags) return;
  const int c = (int)(frag / ntiles), t = (int)(frag % ntiles);
  src += (long)blockIdx.z * srcStride;
  dst += (long)blockIdx.z * dstStride;
  const int kbase = c * 32 + ((lane >> 4) << 4);
  const int n = t * 16 + (lane & 15);
  v16bf v;
#pragma unroll
  for (int j = 0; j < 16; ++j)
    v[j] = (bf16_t)(src[(long)(kbase + j) * N + n] * scale);
  *(v16bf*)(dst + (frag * 32 + lane) * 16) = v;
}

// ----------------------------------------------------------------------------
// Kernel 1: QKV projection + RoPE.
// grid (512 token-tiles, H). 12 waves: waves 0-3 q, 4-7 k, 8-11 v (16 cols each).
// The 16x1024 bf16 activation tile is staged into LDS with
// GLOBAL_LOAD_ASYNC_TO_LDS_B128 (ASYNCcnt path) - no VGPR round-trip.
// Q,K stored row-major (s,64) bf16; V stored transposed (dh,s) bf16.
// 1/sqrt(K) pre-folded into packed wq.
// ----------------------------------------------------------------------------
__global__ __launch_bounds__(384) void qkv_rope_kernel(
    const bf16_t* __restrict__ ebf, const float* __restrict__ cosb,
    const float* __restrict__ sinb,
    const bf16_t* __restrict__ wq_p, const bf16_t* __restrict__ wk_p,
    const bf16_t* __restrict__ wv_p,
    bf16_t* __restrict__ Qg, bf16_t* __restrict__ Kg, bf16_t* __restrict__ Vt) {
  __shared__ bf16_t sX[16 * 1032];          // 16 x E, padded rows (129 x 16B pitch)
  __shared__ float  sQK[2][16][68];         // raw q,k before RoPE
  const int tile = blockIdx.x;
  const int h    = blockIdx.y;
  const int tid  = threadIdx.x;
  const int lane = tid & 31, wave = tid >> 5;
  const long tok0 = (long)tile * 16;
  const int b  = (int)(tok0 / Sc);
  const int s0 = (int)(tok0 - (long)b * Sc);

  // async stage: 16 rows x 128 segments of 16B each, straight into LDS
  {
    const bf16_t* src = ebf + tok0 * E;
    const unsigned ldsBase = (unsigned)(size_t)(&sX[0]);   // low 32 bits = LDS offset
    for (int seg = tid; seg < 16 * 128; seg += 384) {
      const int r = seg >> 7, c8 = seg & 127;
      unsigned ldsAddr = ldsBase + (unsigned)(r * 1032 + c8 * 8) * 2u;
      const bf16_t* g = src + (long)r * E + c8 * 8;
      asm volatile("global_load_async_to_lds_b128 %0, %1, off"
                   :: "v"(ldsAddr), "v"(g) : "memory");
    }
    asm volatile("s_wait_asynccnt 0x0" ::: "memory");
  }
  __syncthreads();

  const int mat = wave >> 2;          // 0=q 1=k 2=v
  const int nt  = wave & 3;
  const bf16_t* wp = (mat == 0) ? wq_p : (mat == 1 ? wk_p : wv_p);
  wp += (long)h * 65536;              // per-head packed elems

  v8f acc = {};
#pragma unroll 4
  for (int c = 0; c < E / 32; ++c) {
    v16bf a = load_a(sX + c * 32, 1032, lane);
    v16bf w = load_b_packed(wp, c * 4 + nt, lane);
    acc = wmma_bf16(a, w, acc);
  }

  const int m0 = (lane >> 4) << 3;
  const int nn = lane & 15;
  if (mat < 2) {
#pragma unroll
    for (int i = 0; i < 8; ++i) sQK[mat][m0 + i][nt * 16 + nn] = acc[i];
  } else {
    v8bf st;
#pragma unroll
    for (int i = 0; i < 8; ++i) st[i] = (bf16_t)acc[i];
    long base = (((long)(b * H + h)) * DH + nt * 16 + nn) * Sc + s0 + m0;
    *(v8bf*)(Vt + base) = st;         // contiguous in s
  }
  __syncthreads();

  // RoPE: out = cos*x + sin*x[(k+32)%64]; write bf16 (s,64)
  for (int idx = tid; idx < 2 * 16 * 64; idx += 384) {
    int qk = idx >> 10;
    int rm = (idx >> 6) & 15;
    int kd = idx & 63;
    float x  = sQK[qk][rm][kd];
    float xs = sQK[qk][rm][(kd + 32) & 63];
    int s = s0 + rm;
    float o = cosb[s * 64 + kd] * x + sinb[s * 64 + kd] * xs;
    long base = (((long)(b * H + h)) * Sc + s) * 64 + kd;
    (qk == 0 ? Qg : Kg)[base] = (bf16_t)o;
  }
}

// ----------------------------------------------------------------------------
// Kernel 2: causal flash attention.  grid (S/16, B*H), 4 waves.
// 64-key chunks: every wave computes one 16x16 score tile (2 WMMAs), the
// online softmax is spread over 32 lanes (two half-rows), then every wave
// runs two P.V WMMAs for its own dh tile (wave w owns cols [16w,16w+16)).
// ----------------------------------------------------------------------------
__global__ __launch_bounds__(128) void attn_kernel(
    const bf16_t* __restrict__ Qg, const bf16_t* __restrict__ Kg,
    const bf16_t* __restrict__ Vt, bf16_t* __restrict__ attn) {
  __shared__ float sc[16][68];
  __shared__ float rowmax[16], rowsum[16], rescale[16];
  __shared__ float hmax[2][16], hsum[2][16];
  const int qt = blockIdx.x;
  const int bh = blockIdx.y;
  const int h  = bh & (H - 1);
  const int b  = bh >> 4;
  const int tid = threadIdx.x, lane = tid & 31, wave = tid >> 5;
  const bf16_t* Qb = Qg + ((long)bh * Sc + qt * 16) * 64;
  const bf16_t* Kb = Kg + (long)bh * Sc * 64;
  const bf16_t* Vb = Vt + (long)bh * DH * Sc;

  if (tid < 16) { rowmax[tid] = -3.0e9f; rowsum[tid] = 0.f; }

  v16bf aq0 = load_a(Qb, 64, lane);
  v16bf aq1 = load_a(Qb + 32, 64, lane);
  v8f accO = {};
  const int m0 = (lane >> 4) << 3;
  const int nn = lane & 15;
  const int nch = (qt * 16 + 79) / 64;   // 64-key chunks covering keys <= qt*16+15

  for (int kc = 0; kc < nch; ++kc) {
    __syncthreads();
    {   // scores: wave w -> keys [kc*64+16w, +16)
      const int key0 = kc * 64 + wave * 16;
      v16bf bk0 = load_b_trans(Kb + (long)key0 * 64, 64, lane);
      v16bf bk1 = load_b_trans(Kb + (long)key0 * 64 + 32, 64, lane);
      v8f s8 = {};
      s8 = wmma_bf16(aq0, bk0, s8);
      s8 = wmma_bf16(aq1, bk1, s8);
#pragma unroll
      for (int i = 0; i < 8; ++i) {
        int srow = qt * 16 + m0 + i;
        int kcol = key0 + nn;
        float v = s8[i];
        if (kcol > srow) v = -1.0e9f;     // causal mask (== buffer values)
        sc[m0 + i][wave * 16 + nn] = v;
      }
    }
    __syncthreads();
    if (tid < 32) {                        // per-half row max
      const int r = tid & 15, hh = tid >> 4;
      float m = -3.0e9f;
#pragma unroll 8
      for (int t = 0; t < 32; ++t) m = fmaxf(m, sc[r][hh * 32 + t]);
      hmax[hh][r] = m;
    }
    __syncthreads();
    if (tid < 16) {                        // combine + rescale factor
      float mx = rowmax[tid];
      float nm = fmaxf(mx, fmaxf(hmax[0][tid], hmax[1][tid]));
      rescale[tid] = __expf(mx - nm);
      rowmax[tid]  = nm;
    }
    __syncthreads();
    if (tid < 32) {                        // exponentiate + per-half sums
      const int r = tid & 15, hh = tid >> 4;
      const float nm = rowmax[r];
      float s = 0.f;
#pragma unroll 8
      for (int t = 0; t < 32; ++t) {
        float p = __expf(sc[r][hh * 32 + t] - nm);
        sc[r][hh * 32 + t] = p;
        s += p;
      }
      hsum[hh][r] = s;
    }
    __syncthreads();
    if (tid < 16)
      rowsum[tid] = rowsum[tid] * rescale[tid] + hsum[0][tid] + hsum[1][tid];

    // P (16x64) x V-chunk (64x16) for this wave's dh tile: 2 WMMAs
    v16bf pa0, pa1;
    {
      const int pm = lane & 15;
      const int kb = (lane >> 4) << 4;
#pragma unroll
      for (int j = 0; j < 8; ++j) {
        pa0[j]     = (bf16_t)sc[pm][kb + j];
        pa0[j + 8] = (bf16_t)sc[pm][kb + 8 + j];
        pa1[j]     = (bf16_t)sc[pm][32 + kb + j];
        pa1[j + 8] = (bf16_t)sc[pm][32 + kb + 8 + j];
      }
    }
    v16bf bv0 = load_b_trans(Vb + (long)(wave * 16) * Sc + kc * 64, Sc, lane);
    v16bf bv1 = load_b_trans(Vb + (long)(wave * 16) * Sc + kc * 64 + 32, Sc, lane);
    v8f cin;
#pragma unroll
    for (int i = 0; i < 8; ++i) cin[i] = accO[i] * rescale[m0 + i];
    accO = wmma_bf16(pa0, bv0, cin);
    accO = wmma_bf16(pa1, bv1, accO);
  }
  __syncthreads();
#pragma unroll
  for (int i = 0; i < 8; ++i) {
    long tok = (long)b * Sc + qt * 16 + m0 + i;
    attn[tok * E + h * 64 + wave * 16 + nn] = (bf16_t)(accO[i] / rowsum[m0 + i]);
  }
}

// ----------------------------------------------------------------------------
// Kernel 3: x = emb + attn @ Wproj.  grid (512, 2), 8 waves.
// Each wave: 4 n-tiles (64 cols) -> one A fragment feeds 4 WMMAs.
// ----------------------------------------------------------------------------
__global__ __launch_bounds__(256) void proj_kernel(
    const bf16_t* __restrict__ attn, const bf16_t* __restrict__ wp_p,
    const float* __restrict__ emb, float* __restrict__ x) {
  const int mt = blockIdx.x;
  const int lane = threadIdx.x & 31, wave = threadIdx.x >> 5;
  const int g = blockIdx.y * 8 + wave;        // 0..15 -> n-tiles 4g..4g+3
  const bf16_t* arow = attn + (long)mt * 16 * E;
  v8f acc0 = {}, acc1 = {}, acc2 = {}, acc3 = {};
#pragma unroll 2
  for (int c = 0; c < E / 32; ++c) {
    v16bf a  = load_a(arow + c * 32, E, lane);
    v16bf b0 = load_b_packed(wp_p, (long)c * 64 + g * 4 + 0, lane);
    v16bf b1 = load_b_packed(wp_p, (long)c * 64 + g * 4 + 1, lane);
    v16bf b2 = load_b_packed(wp_p, (long)c * 64 + g * 4 + 2, lane);
    v16bf b3 = load_b_packed(wp_p, (long)c * 64 + g * 4 + 3, lane);
    acc0 = wmma_bf16(a, b0, acc0);
    acc1 = wmma_bf16(a, b1, acc1);
    acc2 = wmma_bf16(a, b2, acc2);
    acc3 = wmma_bf16(a, b3, acc3);
  }
  const int m0 = (lane >> 4) << 3, nn = lane & 15;
  v8f accs[4] = {acc0, acc1, acc2, acc3};
#pragma unroll
  for (int t = 0; t < 4; ++t) {
    const int col = (g * 4 + t) * 16 + nn;
#pragma unroll
    for (int i = 0; i < 8; ++i) {
      long tok = (long)mt * 16 + m0 + i;
      x[tok * E + col] = emb[tok * E + col] + accs[t][i];
    }
  }
}

// ----------------------------------------------------------------------------
// Kernel 4: h = rmsnorm(x) * w  (bf16 out).  One block per token.
// ----------------------------------------------------------------------------
__global__ __launch_bounds__(256) void rmsnorm_kernel(
    const float* __restrict__ x, const float* __restrict__ w,
    bf16_t* __restrict__ hOut) {
  __shared__ float red[8];
  const long tok = blockIdx.x;
  const float* xr = x + tok * E;
  float ss = 0.f;
  for (int c = threadIdx.x; c < E; c += 256) { float v = xr[c]; ss += v * v; }
#pragma unroll
  for (int off = 16; off; off >>= 1) ss += __shfl_down(ss, off, 32);
  const int lane = threadIdx.x & 31, wave = threadIdx.x >> 5;
  if (lane == 0) red[wave] = ss;
  __syncthreads();
  if (threadIdx.x == 0) {
    float t = 0.f;
#pragma unroll
    for (int i = 0; i < 8; ++i) t += red[i];
    red[0] = rsqrtf(t / (float)E + EPS_F);
  }
  __syncthreads();
  const float r = red[0];
  for (int c = threadIdx.x; c < E; c += 256)
    hOut[tok * E + c] = (bf16_t)(xr[c] * r * w[c]);
}

// ----------------------------------------------------------------------------
// Kernel 5: act = (h@Wup) * elu(h@Wgate).  grid (512, 16), 8 waves.
// Each wave: 2 n-tiles x {up,gate} -> one A fragment feeds 4 WMMAs.
// ----------------------------------------------------------------------------
__global__ __launch_bounds__(256) void mlp_up_kernel(
    const bf16_t* __restrict__ hIn, const bf16_t* __restrict__ wup_p,
    const bf16_t* __restrict__ wgate_p, bf16_t* __restrict__ act) {
  const int mt = blockIdx.x;
  const int lane = threadIdx.x & 31, wave = threadIdx.x >> 5;
  const int g = blockIdx.y * 8 + wave;        // 0..127 -> n-tiles 2g, 2g+1
  const bf16_t* arow = hIn + (long)mt * 16 * E;
  v8f u0 = {}, u1 = {}, g0 = {}, g1 = {};
#pragma unroll 2
  for (int c = 0; c < E / 32; ++c) {
    v16bf a   = load_a(arow + c * 32, E, lane);
    v16bf bu0 = load_b_packed(wup_p,   (long)c * 256 + 2 * g + 0, lane);
    v16bf bu1 = load_b_packed(wup_p,   (long)c * 256 + 2 * g + 1, lane);
    v16bf bg0 = load_b_packed(wgate_p, (long)c * 256 + 2 * g + 0, lane);
    v16bf bg1 = load_b_packed(wgate_p, (long)c * 256 + 2 * g + 1, lane);
    u0 = wmma_bf16(a, bu0, u0);
    u1 = wmma_bf16(a, bu1, u1);
    g0 = wmma_bf16(a, bg0, g0);
    g1 = wmma_bf16(a, bg1, g1);
  }
  const int m0 = (lane >> 4) << 3, nn = lane & 15;
  v8f us[2] = {u0, u1}, gs[2] = {g0, g1};
#pragma unroll
  for (int t = 0; t < 2; ++t) {
    const int col = (2 * g + t) * 16 + nn;
#pragma unroll
    for (int i = 0; i < 8; ++i) {
      float gv = gs[t][i];
      float e = gv > 0.f ? gv : (__expf(gv) - 1.f);   // elu
      long tok = (long)mt * 16 + m0 + i;
      act[tok * F + col] = (bf16_t)(us[t][i] * e);
    }
  }
}

// ----------------------------------------------------------------------------
// Kernel 6: out = x + act @ Wdown.  grid (512, 2), 8 waves.
// Each wave: 4 n-tiles; A rows prefetched 16 k-steps ahead (64 MB stream).
// ----------------------------------------------------------------------------
__global__ __launch_bounds__(256) void down_kernel(
    const bf16_t* __restrict__ act, const bf16_t* __restrict__ wdown_p,
    const float* __restrict__ x, float* __restrict__ out) {
  const int mt = blockIdx.x;
  const int lane = threadIdx.x & 31, wave = threadIdx.x >> 5;
  const int g = blockIdx.y * 8 + wave;        // 0..15 -> n-tiles 4g..4g+3
  const bf16_t* arow = act + (long)mt * 16 * F;
  v8f acc0 = {}, acc1 = {}, acc2 = {}, acc3 = {};
#pragma unroll 2
  for (int c = 0; c < F / 32; ++c) {
    if (c + 16 < F / 32)                       // stream-ahead into caches
      __builtin_prefetch(arow + (long)(lane & 15) * F + (c + 16) * 32, 0, 3);
    v16bf a  = load_a(arow + c * 32, F, lane);
    v16bf b0 = load_b_packed(wdown_p, (long)c * 64 + g * 4 + 0, lane);
    v16bf b1 = load_b_packed(wdown_p, (long)c * 64 + g * 4 + 1, lane);
    v16bf b2 = load_b_packed(wdown_p, (long)c * 64 + g * 4 + 2, lane);
    v16bf b3 = load_b_packed(wdown_p, (long)c * 64 + g * 4 + 3, lane);
    acc0 = wmma_bf16(a, b0, acc0);
    acc1 = wmma_bf16(a, b1, acc1);
    acc2 = wmma_bf16(a, b2, acc2);
    acc3 = wmma_bf16(a, b3, acc3);
  }
  const int m0 = (lane >> 4) << 3, nn = lane & 15;
  v8f accs[4] = {acc0, acc1, acc2, acc3};
#pragma unroll
  for (int t = 0; t < 4; ++t) {
    const int col = (g * 4 + t) * 16 + nn;
#pragma unroll
    for (int i = 0; i < 8; ++i) {
      long tok = (long)mt * 16 + m0 + i;
      out[tok * E + col] = x[tok * E + col] + accs[t][i];
    }
  }
}

// ----------------------------------------------------------------------------
extern "C" void kernel_launch(void* const* d_in, const int* in_sizes, int n_in,
                              void* d_out, int out_size, void* d_ws, size_t ws_size,
                              hipStream_t stream) {
  const float* emb   = (const float*)d_in[0];
  // d_in[1] = causal_buffer (computed analytically in-kernel)
  const float* cosb  = (const float*)d_in[2];
  const float* sinb  = (const float*)d_in[3];
  const float* wq    = (const float*)d_in[4];
  const float* wk    = (const float*)d_in[5];
  const float* wv    = (const float*)d_in[6];
  const float* wproj = (const float*)d_in[7];
  // d_in[8] = attn_norm_w (unused by reference)
  const float* nw    = (const float*)d_in[9];
  const float* wup   = (const float*)d_in[10];
  const float* wgate = (const float*)d_in[11];
  const float* wdown = (const float*)d_in[12];
  float* out = (float*)d_out;

  char* ws = (char*)d_ws;
  size_t off = 0;
  auto take = [&](size_t bytes) -> char* {
    char* p = ws + off;
    off = (off + bytes + 255) & ~(size_t)255;
    return p;
  };

  bf16_t* wq_p    = (bf16_t*)take((size_t)H * 65536 * sizeof(bf16_t));
  bf16_t* wk_p    = (bf16_t*)take((size_t)H * 65536 * sizeof(bf16_t));
  bf16_t* wv_p    = (bf16_t*)take((size_t)H * 65536 * sizeof(bf16_t));
  bf16_t* wproj_p = (bf16_t*)take((size_t)E * E * sizeof(bf16_t));
  bf16_t* wup_p   = (bf16_t*)take((size_t)E * F * sizeof(bf16_t));
  bf16_t* wgate_p = (bf16_t*)take((size_t)E * F * sizeof(bf16_t));
  bf16_t* wdown_p = (bf16_t*)take((size_t)F * E * sizeof(bf16_t));
  bf16_t* ebf     = (bf16_t*)take((size_t)TOK * E * sizeof(bf16_t));
  bf16_t* Qg      = (bf16_t*)take((size_t)Bc * H * Sc * 64 * sizeof(bf16_t));
  bf16_t* Kg      = (bf16_t*)take((size_t)Bc * H * Sc * 64 * sizeof(bf16_t));
  bf16_t* Vt      = (bf16_t*)take((size_t)Bc * H * DH * Sc * sizeof(bf16_t));
  bf16_t* attnb   = (bf16_t*)take((size_t)TOK * E * sizeof(bf16_t));
  float*  xbuf    = (float*)take((size_t)TOK * E * sizeof(float));
  bf16_t* hbuf    = (bf16_t*)take((size_t)TOK * E * sizeof(bf16_t));
  bf16_t* act     = (bf16_t*)take((size_t)TOK * F * sizeof(bf16_t));
  (void)ws_size; (void)in_sizes; (void)n_in; (void)out_size;

  const float qscale = 0.125f;   // 1/sqrt(K=64), folded into wq pack (RoPE is linear)
  pack_kernel<<<dim3(16, 1, H), 256, 0, stream>>>(wq, wq_p, E, 64, (long)E * 64, 65536, qscale);
  pack_kernel<<<dim3(16, 1, H), 256, 0, stream>>>(wk, wk_p, E, 64, (long)E * 64, 65536, 1.f);
  pack_kernel<<<dim3(16, 1, H), 256, 0, stream>>>(wv, wv_p, E, 64, (long)E * 64, 65536, 1.f);
  pack_kernel<<<dim3(256, 1, 1), 256, 0, stream>>>(wproj, wproj_p, E, E, 0, 0, 1.f);
  pack_kernel<<<dim3(1024, 1, 1), 256, 0, stream>>>(wup,   wup_p,   E, F, 0, 0, 1.f);
  pack_kernel<<<dim3(1024, 1, 1), 256, 0, stream>>>(wgate, wgate_p, E, F, 0, 0, 1.f);
  pack_kernel<<<dim3(1024, 1, 1), 256, 0, stream>>>(wdown, wdown_p, F, E, 0, 0, 1.f);

  tobf16_kernel<<<dim3((unsigned)((long)TOK * E / (256 * 8))), 256, 0, stream>>>(
      emb, ebf, (long)TOK * E);

  qkv_rope_kernel<<<dim3(TOK / 16, H), 384, 0, stream>>>(ebf, cosb, sinb,
                                                         wq_p, wk_p, wv_p, Qg, Kg, Vt);
  attn_kernel<<<dim3(Sc / 16, Bc * H), 128, 0, stream>>>(Qg, Kg, Vt, attnb);
  proj_kernel<<<dim3(TOK / 16, 2), 256, 0, stream>>>(attnb, wproj_p, emb, xbuf);
  rmsnorm_kernel<<<dim3(TOK), 256, 0, stream>>>(xbuf, nw, hbuf);
  mlp_up_kernel<<<dim3(TOK / 16, 16), 256, 0, stream>>>(hbuf, wup_p, wgate_p, act);
  down_kernel<<<dim3(TOK / 16, 2), 256, 0, stream>>>(act, wdown_p, xbuf, out);
}
